// MultiHeadAttention_56478819942460
// MI455X (gfx1250) — compile-verified, matrix operand order checked by reading, not run on hardware
//
#include <hip/hip_runtime.h>
#include <hip/hip_bf16.h>

// ---------------- types ----------------
typedef __attribute__((ext_vector_type(16))) __bf16 v16bf;
typedef __attribute__((ext_vector_type(8)))  __bf16 v8bf;
typedef __attribute__((ext_vector_type(8)))  float  v8f;

#define B_  2
#define T_  2048
#define C_  1024
#define H_  16
#define KVH_ 4
#define D_  64

// -------- WMMA helper: D = A(16x32 bf16) * B(32x16 bf16) + C (f32) --------
__device__ __forceinline__ v8f wmma_bf16(v16bf a, v16bf b, v8f c) {
  return __builtin_amdgcn_wmma_f32_16x16x32_bf16(
      /*neg_a=*/false, a, /*neg_b=*/false, b,
      /*c_mod=*/(short)0, c, /*reuse_a=*/false, /*reuse_b=*/false);
}

// Load a 16x32 bf16 A/B fragment from a row-major tile (row stride = ld elems).
// Per ISA 16-bit A layout: lane (g=lane>>4, m=lane&15); elements 0..7 hold
// K = 8g..8g+7, elements 8..15 hold K = 16+8g..23+8g  -> two b128 loads.
__device__ __forceinline__ v16bf load_ab_frag(const __bf16* p, int ld) {
  int lane = threadIdx.x & 31;
  int g = lane >> 4, m = lane & 15;
  const __bf16* q = p + (size_t)m * ld + 8 * g;
  union { v16bf v; v8bf h[2]; } u;
  u.h[0] = *(const v8bf*)(q);
  u.h[1] = *(const v8bf*)(q + 16);
  return u.v;
}

__device__ __forceinline__ float rowmax16(float v) {
  v = fmaxf(v, __shfl_xor(v, 1, 32));
  v = fmaxf(v, __shfl_xor(v, 2, 32));
  v = fmaxf(v, __shfl_xor(v, 4, 32));
  v = fmaxf(v, __shfl_xor(v, 8, 32));
  return v;
}
__device__ __forceinline__ float rowsum16(float v) {
  v += __shfl_xor(v, 1, 32);
  v += __shfl_xor(v, 2, 32);
  v += __shfl_xor(v, 4, 32);
  v += __shfl_xor(v, 8, 32);
  return v;
}

// ---------------- prep kernels ----------------
__global__ void cvt_f32_bf16(const float* __restrict__ src, __bf16* __restrict__ dst, int n) {
  int i = blockIdx.x * blockDim.x + threadIdx.x;
  if (i < n) dst[i] = (__bf16)src[i];
}

// W is KxN row-major f32; write Wt = NxK row-major bf16
__global__ void transpose_cvt(const float* __restrict__ W, __bf16* __restrict__ Wt, int K, int N) {
  int i = blockIdx.x * blockDim.x + threadIdx.x;
  if (i >= K * N) return;
  int k = i / N, n = i - k * N;
  Wt[(size_t)n * K + k] = (__bf16)W[i];
}

// ---------------- WMMA GEMM: D[MxN] = A[MxK] * Bt[NxK]^T ----------------
// block = 128 threads (4 waves). wave w: rows m0..m0+15, cols n0..n0+63 (4 tiles)
__global__ __launch_bounds__(128) void gemm_bf16_nt(const __bf16* __restrict__ A,
                                                    const __bf16* __restrict__ Bt,
                                                    float* __restrict__ D,
                                                    int M, int N, int K) {
  int w = threadIdx.x >> 5;
  int lane = threadIdx.x & 31;
  int g = lane >> 4, n = lane & 15;
  int m0 = blockIdx.y * 16;
  int n0 = blockIdx.x * 256 + w * 64;
  if (m0 >= M || n0 >= N) return;  // wave-uniform
  v8f zero = {0.f, 0.f, 0.f, 0.f, 0.f, 0.f, 0.f, 0.f};
  v8f acc0 = zero, acc1 = zero, acc2 = zero, acc3 = zero;
  const __bf16* arow = A + (size_t)m0 * K;
  for (int k0 = 0; k0 < K; k0 += 32) {
    v16bf a = load_ab_frag(arow + k0, K);
    v16bf b0 = load_ab_frag(Bt + (size_t)(n0 + 0)  * K + k0, K);
    acc0 = wmma_bf16(a, b0, acc0);
    v16bf b1 = load_ab_frag(Bt + (size_t)(n0 + 16) * K + k0, K);
    acc1 = wmma_bf16(a, b1, acc1);
    v16bf b2 = load_ab_frag(Bt + (size_t)(n0 + 32) * K + k0, K);
    acc2 = wmma_bf16(a, b2, acc2);
    v16bf b3 = load_ab_frag(Bt + (size_t)(n0 + 48) * K + k0, K);
    acc3 = wmma_bf16(a, b3, acc3);
  }
#pragma unroll
  for (int r = 0; r < 8; ++r) {
    size_t row = (size_t)(m0 + r + 8 * g);
    float* drow = D + row * N + n0 + n;
    drow[0]  = acc0[r];
    drow[16] = acc1[r];
    drow[32] = acc2[r];
    drow[48] = acc3[r];
  }
}

// ---------------- RoPE + packing ----------------
#define LOG_ROPE_BASE 9.210340371976184f  // ln(10000)

// q: f32 [B*T, C] -> bf16 [B,H,T,D] with RoPE
__global__ void rope_pack_q(const float* __restrict__ qt, __bf16* __restrict__ qbf) {
  int id = blockIdx.x * blockDim.x + threadIdx.x;  // ((b*T+t)*H+h)*32+i
  if (id >= B_ * T_ * H_ * 32) return;
  int i = id & 31;
  int h = (id >> 5) & (H_ - 1);
  int t = (id >> 9) & (T_ - 1);
  int b = id >> 20;
  const float* src = qt + ((size_t)(b * T_ + t) * C_) + h * D_ + 2 * i;
  float x1 = src[0], x2 = src[1];
  float inv = __expf(-(2.0f * i / (float)D_) * LOG_ROPE_BASE);
  float ang = (float)t * inv;
  float c = __cosf(ang), s = __sinf(ang);
  __bf16* dst = qbf + ((size_t)(b * H_ + h) * T_ + t) * D_ + 2 * i;
  dst[0] = (__bf16)(x1 * c - x2 * s);
  dst[1] = (__bf16)(x1 * s + x2 * c);
}

// k: f32 [B*T, KVH*D] -> bf16 [B,KVH,T,D] with RoPE
__global__ void rope_pack_k(const float* __restrict__ kt, __bf16* __restrict__ kbf) {
  int id = blockIdx.x * blockDim.x + threadIdx.x;  // ((b*T+t)*KVH+h)*32+i
  if (id >= B_ * T_ * KVH_ * 32) return;
  int i = id & 31;
  int h = (id >> 5) & (KVH_ - 1);
  int t = (id >> 7) & (T_ - 1);
  int b = id >> 18;
  const float* src = kt + ((size_t)(b * T_ + t) * (KVH_ * D_)) + h * D_ + 2 * i;
  float x1 = src[0], x2 = src[1];
  float inv = __expf(-(2.0f * i / (float)D_) * LOG_ROPE_BASE);
  float ang = (float)t * inv;
  float c = __cosf(ang), s = __sinf(ang);
  __bf16* dst = kbf + ((size_t)(b * KVH_ + h) * T_ + t) * D_ + 2 * i;
  dst[0] = (__bf16)(x1 * c - x2 * s);
  dst[1] = (__bf16)(x1 * s + x2 * c);
}

// v: f32 [B*T, KVH*D] -> bf16 transposed [B,KVH,D,T]
__global__ void pack_v(const float* __restrict__ vt, __bf16* __restrict__ vtbf) {
  int id = blockIdx.x * blockDim.x + threadIdx.x;  // (b*T+t)*256 + c
  if (id >= B_ * T_ * KVH_ * D_) return;
  int c = id & (KVH_ * D_ - 1);
  int t = (id >> 8) & (T_ - 1);
  int b = id >> 19;
  int h = c >> 6, d = c & (D_ - 1);
  vtbf[((size_t)(b * KVH_ + h) * D_ + d) * T_ + t] = (__bf16)vt[id];
}

// ---------------- flash attention (mask-times-score semantics) ----------------
// one wave per (b, h, 16-query tile). Streams ALL keys; masked scores = 0
// (reference multiplies the mask BEFORE softmax, so masked entries contribute
// exp(0-max) to the denominator and pull in V rows — reproduced exactly).
__global__ __launch_bounds__(128) void attn_kernel(const __bf16* __restrict__ qbf,
                                                   const __bf16* __restrict__ kbf,
                                                   const __bf16* __restrict__ vtbf,
                                                   __bf16* __restrict__ attnbf) {
  __shared__ __bf16 plds_all[4][16 * 32];
  int w = threadIdx.x >> 5;
  int lane = threadIdx.x & 31;
  int g = lane >> 4, n = lane & 15;
  int wid = blockIdx.x * 4 + w;
  int qt = wid & 127;           // T/16
  int h = (wid >> 7) & (H_ - 1);
  int b = wid >> 11;
  int kvh = h & (KVH_ - 1);     // jnp.tile => h % N_KV_HEAD
  int m0 = qt * 16;

  const __bf16* qbase = qbf + ((size_t)(b * H_ + h) * T_ + m0) * D_;
  v16bf qf0 = load_ab_frag(qbase + 0, D_);
  v16bf qf1 = load_ab_frag(qbase + 32, D_);

  v8f zero = {0.f, 0.f, 0.f, 0.f, 0.f, 0.f, 0.f, 0.f};
  v8f o0 = zero, o1 = zero, o2 = zero, o3 = zero;
  float mprev[8], lsum[8];
#pragma unroll
  for (int r = 0; r < 8; ++r) { mprev[r] = -1e30f; lsum[r] = 0.f; }

  __bf16* plds = &plds_all[w][0];
  const __bf16* kh = kbf + (size_t)(b * KVH_ + kvh) * T_ * D_;
  const __bf16* vh = vtbf + (size_t)(b * KVH_ + kvh) * D_ * T_;

  for (int j0 = 0; j0 < T_; j0 += 32) {
    // ---- S = Q * K^T for 32 keys (two 16x16 tiles) ----
    v8f s0 = zero, s1 = zero;
    {
      v16bf kf0 = load_ab_frag(kh + (size_t)j0 * D_ + 0, D_);
      v16bf kf1 = load_ab_frag(kh + (size_t)j0 * D_ + 32, D_);
      s0 = wmma_bf16(qf0, kf0, s0);
      s0 = wmma_bf16(qf1, kf1, s0);
      v16bf kf2 = load_ab_frag(kh + (size_t)(j0 + 16) * D_ + 0, D_);
      v16bf kf3 = load_ab_frag(kh + (size_t)(j0 + 16) * D_ + 32, D_);
      s1 = wmma_bf16(qf0, kf2, s1);
      s1 = wmma_bf16(qf1, kf3, s1);
    }
    // ---- online softmax per row (D-layout: row = r+8g, col = lane n) ----
#pragma unroll
    for (int r = 0; r < 8; ++r) {
      int qi = m0 + r + 8 * g;
      float v0 = s0[r] * 0.125f;                 // 1/sqrt(64)
      v0 = (j0 + n > qi) ? 0.f : v0;             // mask*att (zero, not -inf)
      float v1 = s1[r] * 0.125f;
      v1 = (j0 + 16 + n > qi) ? 0.f : v1;
      float cm = rowmax16(fmaxf(v0, v1));
      float mn = fmaxf(mprev[r], cm);
      float corr = __expf(mprev[r] - mn);
      float p0 = __expf(v0 - mn);
      float p1 = __expf(v1 - mn);
      float ps = rowsum16(p0 + p1);
      lsum[r] = lsum[r] * corr + ps;
      mprev[r] = mn;
      o0[r] = o0[r] * corr;
      o1[r] = o1[r] * corr;
      o2[r] = o2[r] * corr;
      o3[r] = o3[r] * corr;
      // stash P (D-layout -> LDS row-major 16x32 for A-fragment reload)
      plds[(r + 8 * g) * 32 + n] = (__bf16)p0;
      plds[(r + 8 * g) * 32 + 16 + n] = (__bf16)p1;
    }
    // ---- O += P * V (contraction over 32 keys, 4 d-tiles) ----
    v16bf pf = load_ab_frag(plds, 32);
    v16bf vf0 = load_ab_frag(vh + (size_t)(0 * 16) * T_ + j0, T_);
    o0 = wmma_bf16(pf, vf0, o0);
    v16bf vf1 = load_ab_frag(vh + (size_t)(1 * 16) * T_ + j0, T_);
    o1 = wmma_bf16(pf, vf1, o1);
    v16bf vf2 = load_ab_frag(vh + (size_t)(2 * 16) * T_ + j0, T_);
    o2 = wmma_bf16(pf, vf2, o2);
    v16bf vf3 = load_ab_frag(vh + (size_t)(3 * 16) * T_ + j0, T_);
    o3 = wmma_bf16(pf, vf3, o3);
  }

  // ---- normalize and write [B,T,C] bf16 for the output projection ----
#pragma unroll
  for (int r = 0; r < 8; ++r) {
    float inv = 1.0f / lsum[r];
    size_t row = (size_t)b * T_ + m0 + r + 8 * g;
    __bf16* dst = attnbf + row * C_ + h * D_ + n;
    dst[0]  = (__bf16)(o0[r] * inv);
    dst[16] = (__bf16)(o1[r] * inv);
    dst[32] = (__bf16)(o2[r] * inv);
    dst[48] = (__bf16)(o3[r] * inv);
  }
}

// ---------------- host-side launch ----------------
extern "C" void kernel_launch(void* const* d_in, const int* in_sizes, int n_in,
                              void* d_out, int out_size, void* d_ws, size_t ws_size,
                              hipStream_t stream) {
  (void)in_sizes; (void)n_in; (void)out_size; (void)ws_size;
  const float* x  = (const float*)d_in[0];
  const float* Wq = (const float*)d_in[1];
  const float* Wk = (const float*)d_in[2];
  const float* Wv = (const float*)d_in[3];
  const float* Wo = (const float*)d_in[4];
  float* out = (float*)d_out;

  char* wsp = (char*)d_ws;
  const size_t MB = 1u << 20;
  __bf16* xb     = (__bf16*)(wsp + 0);          // 8 MiB  [B*T, C]
  __bf16* WqT    = (__bf16*)(wsp + 8 * MB);     // 2 MiB  [C, C]
  __bf16* WkT    = (__bf16*)(wsp + 10 * MB);    // 0.5    [256, C]
  __bf16* WvT    = (__bf16*)(wsp + 10 * MB + 512 * 1024);  // 0.5
  __bf16* WoT    = (__bf16*)(wsp + 11 * MB);    // 2 MiB  [C, C]
  float*  qtmp   = (float*)(wsp + 13 * MB);     // 16 MiB [B*T, C]
  float*  ktmp   = (float*)(wsp + 29 * MB);     // 4 MiB  [B*T, 256]
  float*  vtmp   = (float*)(wsp + 33 * MB);     // 4 MiB  [B*T, 256]
  __bf16* qbf    = (__bf16*)(wsp + 37 * MB);    // 8 MiB  [B,H,T,D]
  __bf16* kbf    = (__bf16*)(wsp + 45 * MB);    // 2 MiB  [B,KVH,T,D]
  __bf16* vtbf   = (__bf16*)(wsp + 47 * MB);    // 2 MiB  [B,KVH,D,T]
  __bf16* attnbf = (__bf16*)(wsp + 49 * MB);    // 8 MiB  [B*T, C]

  const int M = B_ * T_;  // 4096

  // 1. casts / transposes
  cvt_f32_bf16<<<(M * C_ + 255) / 256, 256, 0, stream>>>(x, xb, M * C_);
  transpose_cvt<<<(C_ * C_ + 255) / 256, 256, 0, stream>>>(Wq, WqT, C_, C_);
  transpose_cvt<<<(C_ * KVH_ * D_ + 255) / 256, 256, 0, stream>>>(Wk, WkT, C_, KVH_ * D_);
  transpose_cvt<<<(C_ * KVH_ * D_ + 255) / 256, 256, 0, stream>>>(Wv, WvT, C_, KVH_ * D_);
  transpose_cvt<<<(C_ * C_ + 255) / 256, 256, 0, stream>>>(Wo, WoT, C_, C_);

  // 2. QKV projections (WMMA)
  gemm_bf16_nt<<<dim3(C_ / 256, M / 16), 128, 0, stream>>>(xb, WqT, qtmp, M, C_, C_);
  gemm_bf16_nt<<<dim3(1, M / 16), 128, 0, stream>>>(xb, WkT, ktmp, M, KVH_ * D_, C_);
  gemm_bf16_nt<<<dim3(1, M / 16), 128, 0, stream>>>(xb, WvT, vtmp, M, KVH_ * D_, C_);

  // 3. RoPE + pack
  rope_pack_q<<<(B_ * T_ * H_ * 32 + 255) / 256, 256, 0, stream>>>(qtmp, qbf);
  rope_pack_k<<<(B_ * T_ * KVH_ * 32 + 255) / 256, 256, 0, stream>>>(ktmp, kbf);
  pack_v<<<(B_ * T_ * KVH_ * D_ + 255) / 256, 256, 0, stream>>>(vtmp, vtbf);

  // 4. attention (WMMA flash loop): 4096 waves, 4 waves/block
  attn_kernel<<<(B_ * H_ * (T_ / 16)) / 4, 128, 0, stream>>>(qbf, kbf, vtbf, attnbf);

  // 5. output projection (WMMA) -> fp32 d_out
  gemm_bf16_nt<<<dim3(C_ / 256, M / 16), 128, 0, stream>>>(attnbf, WoT, out, M, C_, C_);
}
